// model_encdec_8770323218756
// MI455X (gfx1250) — matrix operand order repaired and checked by reading, not compiled, not run on hardware
//
#include <hip/hip_runtime.h>
#include <stdint.h>

// ---------------- WMMA types / helpers (CDNA5 gfx1250, wave32) ----------------
typedef __bf16 v16bf __attribute__((ext_vector_type(16)));
typedef float  v8f   __attribute__((ext_vector_type(8)));

union BFrag { v16bf v; uint32_t u[8]; uint4 q[2]; };

__device__ __forceinline__ uint16_t f2bf(float f) {            // RNE f32 -> bf16
  uint32_t x = __float_as_uint(f);
  return (uint16_t)((x + 0x7fffu + ((x >> 16) & 1u)) >> 16);
}
__device__ __forceinline__ uint32_t pk2(float lo, float hi) {
  return (uint32_t)f2bf(lo) | ((uint32_t)f2bf(hi) << 16);
}
__device__ __forceinline__ float bflo(uint32_t pe) { return __uint_as_float(pe << 16); }
__device__ __forceinline__ float bfhi(uint32_t pe) { return __uint_as_float(pe & 0xffff0000u); }

__device__ __forceinline__ v8f z8() { v8f c = {0.f,0.f,0.f,0.f,0.f,0.f,0.f,0.f}; return c; }

__device__ __forceinline__ v8f wmma_bf16(v16bf a, v16bf b, v8f c) {
  return __builtin_amdgcn_wmma_f32_16x16x32_bf16(false, a, false, b, (short)0, c, false, false);
}

// A-frag from PRE-PACKED bf16 LDS (row r at base[r*rstride], 16B-aligned runs).
// lane<16 -> K kb+0..7 (u0..3), kb+16..23 (u4..7); lane>=16 -> kb+8..15, kb+24..31.
__device__ __forceinline__ v16bf afrag_full(const uint16_t* base, int rstride, int lane, int kbase) {
  int r = lane & 15, half = lane >> 4;
  const uint16_t* p = base + r * rstride + kbase + half * 8;
  BFrag f;
  f.q[0] = *(const uint4*)p;          // one ds_load_b128
  f.q[1] = *(const uint4*)(p + 16);   // one ds_load_b128
  return f.v;
}
// Upper 16 K-values of the chunk are beyond kvalid -> hard zero (no guards).
__device__ __forceinline__ v16bf afrag_half(const uint16_t* base, int rstride, int lane, int kbase) {
  int r = lane & 15, half = lane >> 4;
  const uint16_t* p = base + r * rstride + kbase + half * 8;
  BFrag f;
  f.q[0] = *(const uint4*)p;
  f.u[4] = f.u[5] = f.u[6] = f.u[7] = 0u;
  return f.v;
}

// B-frag: B[k][n] = W[nb+n][k], W row-major f32 (global, loaded once per phase).
__device__ __forceinline__ v16bf bfrag(const float* W, int ld, int nb, int lane,
                                       int kbase, int kvalid) {
  int n = lane & 15, half = lane >> 4;
  int ks = kbase + half * 8;
  BFrag f;
#pragma unroll
  for (int i = 0; i < 8; ++i) {
    int k = ks + 2 * (i & 3) + (i >> 2) * 16;
    float lo = (k     < kvalid) ? W[(nb + n) * ld + k    ] : 0.f;
    float hi = (k + 1 < kvalid) ? W[(nb + n) * ld + k + 1] : 0.f;
    f.u[i] = pk2(lo, hi);
  }
  return f.v;
}

// C/D (16x16 f32) -> LDS f32 buffer [16][ld] at column nb.
__device__ __forceinline__ void cstore(float* buf, int ld, int nb, int lane, v8f c) {
  int n = lane & 15, half = lane >> 4;
#pragma unroll
  for (int i = 0; i < 8; ++i) buf[(i + 8 * half) * ld + nb + n] = c[i];
}

// Fast activations on hardware trans units (v_exp_f32 / v_rcp_f32).
__device__ __forceinline__ float fsigm(float x) {
  return __builtin_amdgcn_rcpf(1.0f + __builtin_amdgcn_exp2f(-1.4426950408889634f * x));
}
__device__ __forceinline__ float ftanh(float x) {
  return 1.0f - 2.0f * __builtin_amdgcn_rcpf(__builtin_amdgcn_exp2f(2.8853900817779268f * x) + 1.0f);
}

// ---------------- LDS arena layout (bytes, all 16B aligned) ----------------
// enc_out  bf16 [16][60][48] :      0            (92160)
// enc_part bf16 [16][60][64] :  92160            (122880)  overlays embeds bf16 [16][60][16]
// h        f32  [16][48]     : 215040            (3072)
// h_bf     bf16 [16][48]     : 218112            (1536)   pre-packed WMMA A copy
// xw       f32  [16][144]    : 219648            (9216)
// hw       f32  [16][144]    : 228864            (9216)
// hpart    f32  [16][64]     : 238080            (4096)
// scores   f32  [16][60]     : 242176            (3840)
// ctx      f32  [16][48]     : 246016            (3072)
// ctx_bf   bf16 [16][48]     : 249088            (1536)
// coords   f32  [16][2]      : 250624            (128)
#define SMEM_BYTES 250752

__global__ __launch_bounds__(128, 1)
void encdec_kernel(const float* __restrict__ past, const float* __restrict__ future,
                   const float* __restrict__ cpw, const float* __restrict__ cpb,
                   const float* __restrict__ cfw, const float* __restrict__ cfb,
                   const float* __restrict__ gp_wih, const float* __restrict__ gp_whh,
                   const float* __restrict__ gp_bih, const float* __restrict__ gp_bhh,
                   const float* __restrict__ gf_wih, const float* __restrict__ gf_whh,
                   const float* __restrict__ gf_bih, const float* __restrict__ gf_bhh,
                   const float* __restrict__ d_wih, const float* __restrict__ d_whh,
                   const float* __restrict__ d_bih, const float* __restrict__ d_bhh,
                   const float* __restrict__ a1w, const float* __restrict__ a1b,
                   const float* __restrict__ a2w, const float* __restrict__ a2b,
                   const float* __restrict__ fcw, const float* __restrict__ fcb,
                   float* __restrict__ out) {
  extern __shared__ char smem[];
  uint16_t* s_enc   = (uint16_t*)(smem);
  uint16_t* s_encp  = (uint16_t*)(smem + 92160);
  uint16_t* s_embbf = (uint16_t*)(smem + 92160);   // overlay, dead before enc_part is written
  float*    s_h     = (float*)   (smem + 215040);
  uint16_t* s_hbf   = (uint16_t*)(smem + 218112);
  float*    s_xw    = (float*)   (smem + 219648);
  float*    s_hw    = (float*)   (smem + 228864);
  float*    s_hp    = (float*)   (smem + 238080);
  float*    s_sc    = (float*)   (smem + 242176);
  float*    s_ctx   = (float*)   (smem + 246016);
  uint16_t* s_ctxbf = (uint16_t*)(smem + 249088);
  float*    s_crd   = (float*)   (smem + 250624);

  const int tid = threadIdx.x;
  const int lane = tid & 31;
  const int wid = tid >> 5;
  const int rowbase = blockIdx.x * 16;

  __builtin_prefetch(d_wih, 0, 1);
  __builtin_prefetch(d_whh, 0, 1);
  __builtin_prefetch(a1w, 0, 1);

  // 9 N-tiles of the 144-wide gate matmuls split over 4 waves.
  const int ntl = (wid == 3) ? 3 : 2;
  const int tls[3] = {wid, wid + 4, 8};

  // -------- Phase 1: conv1d embeds (leaky ReLU) -> packed bf16 --------
  for (int it = 0; it < 120; ++it) {
    int idx = tid + 128 * it;                 // [r][t][co], 16*60*16
    int co = idx & 15, t = (idx >> 4) % 60, r = idx / 960;
    size_t b = (size_t)(rowbase + r);
    float acc; const float* w; const float* xb; int L, tl;
    if (t < 20) { w = cpw; acc = cpb[co]; xb = past + b * 40;   L = 20; tl = t; }
    else        { w = cfw; acc = cfb[co]; xb = future + b * 80; L = 40; tl = t - 20; }
#pragma unroll
    for (int k = 0; k < 3; ++k) {
      int tt = tl + k - 1;
      if (tt >= 0 && tt < L) {
        acc += xb[tt * 2 + 0] * w[(co * 2 + 0) * 3 + k];
        acc += xb[tt * 2 + 1] * w[(co * 2 + 1) * 3 + k];
      }
    }
    s_embbf[idx] = f2bf((acc >= 0.f) ? acc : 0.1f * acc);
  }

  // -------- Phase 2: two GRU encoders --------
  for (int ph = 0; ph < 2; ++ph) {
    const float* wih = ph ? gf_wih : gp_wih;
    const float* whh = ph ? gf_whh : gp_whh;
    const float* bih = ph ? gf_bih : gp_bih;
    const float* bhh = ph ? gf_bhh : gp_bhh;
    const int tbase = ph ? 20 : 0, tlen = ph ? 40 : 20;

#pragma unroll
    for (int i = 0; i < 6; ++i) s_h[tid + 128 * i] = 0.f;
#pragma unroll
    for (int i = 0; i < 3; ++i) ((uint32_t*)s_hbf)[tid + 128 * i] = 0u;
    __syncthreads();

    v16bf bx[3], bh0[3], bh1[3];                               // hoisted B fragments
#pragma unroll
    for (int i = 0; i < 3; ++i) if (i < ntl) {
      int nb = tls[i] * 16;
      bx[i]  = bfrag(wih, 16, nb, lane, 0, 16);
      bh0[i] = bfrag(whh, 48, nb, lane, 0, 48);
      bh1[i] = bfrag(whh, 48, nb, lane, 32, 48);
    }

    for (int tt = 0; tt < tlen; ++tt) {
      int t = tbase + tt;
      v16bf ax  = afrag_half(s_embbf + t * 16, 960, lane, 0);   // K 0..15 valid
      v16bf ah0 = afrag_full(s_hbf, 48, lane, 0);
      v16bf ah1 = afrag_half(s_hbf, 48, lane, 32);              // K 32..47 valid
#pragma unroll
      for (int i = 0; i < 3; ++i) if (i < ntl) {
        v8f cx = wmma_bf16(ax, bx[i], z8());
        cstore(s_xw, 144, tls[i] * 16, lane, cx);
        v8f ch = wmma_bf16(ah0, bh0[i], z8());
        ch = wmma_bf16(ah1, bh1[i], ch);
        cstore(s_hw, 144, tls[i] * 16, lane, ch);
      }
      __syncthreads();
#pragma unroll
      for (int i = 0; i < 6; ++i) {                            // GRU gates
        int idx = tid + 128 * i; int r = idx / 48, d = idx - r * 48;
        float xr = s_xw[r*144 + d]      + bih[d],      hr = s_hw[r*144 + d]      + bhh[d];
        float xz = s_xw[r*144 + 48 + d] + bih[48 + d], hz = s_hw[r*144 + 48 + d] + bhh[48 + d];
        float xn = s_xw[r*144 + 96 + d] + bih[96 + d], hn = s_hw[r*144 + 96 + d] + bhh[96 + d];
        float rg = fsigm(xr + hr), zg = fsigm(xz + hz);
        float ng = ftanh(xn + rg * hn);
        float hnew = (1.f - zg) * ng + zg * s_h[r*48 + d];
        s_h[r*48 + d] = hnew;
        uint16_t hb = f2bf(hnew);
        s_hbf[r*48 + d] = hb;                                  // packed WMMA-A copy
        s_enc[(r*60 + t)*48 + d] = hb;                         // encoder output
      }
      __syncthreads();
    }
  }

  // -------- Phase 3: loop-invariant  enc_part = enc_out @ a1w[:,48:]^T --------
  {
    v16bf bp0[4], bp1[4];
#pragma unroll
    for (int nt = 0; nt < 4; ++nt) {
      bp0[nt] = bfrag(a1w, 96, nt * 16, lane, 48, 96);
      bp1[nt] = bfrag(a1w, 96, nt * 16, lane, 80, 96);
    }
    for (int t = wid; t < 60; t += 4) {
      v16bf a0 = afrag_full(s_enc + t * 48, 2880, lane, 0);
      v16bf a1 = afrag_half(s_enc + t * 48, 2880, lane, 32);
#pragma unroll
      for (int nt = 0; nt < 4; ++nt) {
        v8f c = wmma_bf16(a0, bp0[nt], z8());
        c = wmma_bf16(a1, bp1[nt], c);
        int n = lane & 15, half = lane >> 4;
#pragma unroll
        for (int i = 0; i < 8; ++i)
          s_encp[((i + 8*half)*60 + t)*64 + nt*16 + n] = f2bf(c[i]);
      }
    }
  }
  __syncthreads();

  // -------- Phase 4: attention decoder, 40 steps --------
#pragma unroll
  for (int i = 0; i < 6; ++i) s_h[tid + 128 * i] = 0.f;
#pragma unroll
  for (int i = 0; i < 3; ++i) ((uint32_t*)s_hbf)[tid + 128 * i] = 0u;
  if (tid < 32) {
    int r = tid >> 1, o = tid & 1;
    s_crd[tid] = past[((size_t)(rowbase + r) * 20 + 19) * 2 + o];
  }
  __syncthreads();

  v16bf bhp0 = bfrag(a1w, 96, wid * 16, lane, 0, 48);
  v16bf bhp1 = bfrag(a1w, 96, wid * 16, lane, 32, 48);
  v16bf bdx0[3], bdx1[3], bdh0[3], bdh1[3];
#pragma unroll
  for (int i = 0; i < 3; ++i) if (i < ntl) {
    int nb = tls[i] * 16;
    bdx0[i] = bfrag(d_wih, 48, nb, lane, 0, 48);
    bdx1[i] = bfrag(d_wih, 48, nb, lane, 32, 48);
    bdh0[i] = bfrag(d_whh, 48, nb, lane, 0, 48);
    bdh1[i] = bfrag(d_whh, 48, nb, lane, 32, 48);
  }
  const float a2bias = a2b[0];

  for (int s = 0; s < 40; ++s) {
    // (a) h_part = h @ a1w[:, :48]^T + a1b   (one N-tile per wave)
    {
      v16bf ah0 = afrag_full(s_hbf, 48, lane, 0);
      v16bf ah1 = afrag_half(s_hbf, 48, lane, 32);
      v8f c = wmma_bf16(ah0, bhp0, z8());
      c = wmma_bf16(ah1, bhp1, c);
      int n = lane & 15, half = lane >> 4;
      float ab = a1b[wid * 16 + n];
#pragma unroll
      for (int i = 0; i < 8; ++i) s_hp[(i + 8*half)*64 + wid*16 + n] = c[i] + ab;
    }
    __syncthreads();

    // (b) scores[r][t] = a2 . tanh(h_part[r] + enc_part[r][t])   (paired bf16 loads)
    for (int qi = 0; qi < 8; ++qi) {
      int q = tid + 128 * qi;
      if (q < 960) {
        int t = q >> 4, r = q & 15;
        const float* hp = &s_hp[r * 64];
        const uint16_t* ep = &s_encp[(r * 60 + t) * 64];
        float acc = a2bias;
#pragma unroll 8
        for (int j2 = 0; j2 < 32; ++j2) {
          uint32_t pe = *(const uint32_t*)(ep + 2 * j2);
          acc += a2w[2*j2]     * ftanh(hp[2*j2]     + bflo(pe));
          acc += a2w[2*j2 + 1] * ftanh(hp[2*j2 + 1] + bfhi(pe));
        }
        s_sc[r * 60 + t] = acc;
      }
    }
    __syncthreads();

    // (c) softmax over t=60 (one row per lane)
    if (tid < 16) {
      int r = tid; float m = -1e30f;
      for (int t = 0; t < 60; ++t) m = fmaxf(m, s_sc[r*60 + t]);
      float sum = 0.f;
      for (int t = 0; t < 60; ++t) {
        float e = __builtin_amdgcn_exp2f((s_sc[r*60 + t] - m) * 1.4426950408889634f);
        s_sc[r*60 + t] = e; sum += e;
      }
      float inv = __builtin_amdgcn_rcpf(sum);
      for (int t = 0; t < 60; ++t) s_sc[r*60 + t] *= inv;
    }
    __syncthreads();

    // (d) ctx = sum_t w[r][t] * enc_out[r][t][:]  (2 outputs/thread, paired loads)
#pragma unroll
    for (int i = 0; i < 3; ++i) {
      int idx = tid + 128 * i;                  // 16*24 = 384 pairs
      int r = idx / 24, d2 = idx - r * 24;
      const float* wp = &s_sc[r * 60];
      const uint16_t* e = &s_enc[r * 60 * 48 + 2 * d2];
      float acc0 = 0.f, acc1 = 0.f;
#pragma unroll 4
      for (int t = 0; t < 60; ++t) {
        uint32_t pe = *(const uint32_t*)(e + t * 48);
        float w = wp[t];
        acc0 += w * bflo(pe); acc1 += w * bfhi(pe);
      }
      s_ctx[r*48 + 2*d2] = acc0; s_ctx[r*48 + 2*d2 + 1] = acc1;
      *(uint32_t*)&s_ctxbf[r*48 + 2*d2] = pk2(acc0, acc1);
    }
    __syncthreads();

    // (e) decoder GRU-cell matmuls: xw = ctx @ d_wih^T, hw = h @ d_whh^T
    {
      v16bf ax0 = afrag_full(s_ctxbf, 48, lane, 0);
      v16bf ax1 = afrag_half(s_ctxbf, 48, lane, 32);
      v16bf ah0 = afrag_full(s_hbf, 48, lane, 0);
      v16bf ah1 = afrag_half(s_hbf, 48, lane, 32);
#pragma unroll
      for (int i = 0; i < 3; ++i) if (i < ntl) {
        v8f cx = wmma_bf16(ax0, bdx0[i], z8());
        cx = wmma_bf16(ax1, bdx1[i], cx);
        cstore(s_xw, 144, tls[i] * 16, lane, cx);
        v8f ch = wmma_bf16(ah0, bdh0[i], z8());
        ch = wmma_bf16(ah1, bdh1[i], ch);
        cstore(s_hw, 144, tls[i] * 16, lane, ch);
      }
    }
    __syncthreads();

    // (f) GRU gates -> h_new (+ packed bf16 mirror)
#pragma unroll
    for (int i = 0; i < 6; ++i) {
      int idx = tid + 128 * i; int r = idx / 48, d = idx - r * 48;
      float xr = s_xw[r*144 + d]      + d_bih[d],      hr = s_hw[r*144 + d]      + d_bhh[d];
      float xz = s_xw[r*144 + 48 + d] + d_bih[48 + d], hz = s_hw[r*144 + 48 + d] + d_bhh[48 + d];
      float xn = s_xw[r*144 + 96 + d] + d_bih[96 + d], hn = s_hw[r*144 + 96 + d] + d_bhh[96 + d];
      float rg = fsigm(xr + hr), zg = fsigm(xz + hz);
      float ng = ftanh(xn + rg * hn);
      float hnew = (1.f - zg) * ng + zg * s_h[r*48 + d];
      s_h[r*48 + d] = hnew;
      s_hbf[r*48 + d] = f2bf(hnew);
    }
    __syncthreads();

    // (g) disp = [h_new, ctx] @ fc^T + fc_b ; coords += disp ; emit
    if (tid < 32) {
      int r = tid >> 1, o = tid & 1;
      float acc = fcb[o];
      const float* fw = &fcw[o * 96];
#pragma unroll 8
      for (int d = 0; d < 48; ++d)
        acc += fw[d] * s_h[r*48 + d] + fw[48 + d] * s_ctx[r*48 + d];
      float nc = s_crd[tid] + acc;
      s_crd[tid] = nc;
      out[((size_t)(rowbase + r) * 40 + s) * 2 + o] = nc;
    }
    __syncthreads();
  }
}

extern "C" void kernel_launch(void* const* d_in, const int* in_sizes, int n_in,
                              void* d_out, int out_size, void* d_ws, size_t ws_size,
                              hipStream_t stream) {
  (void)n_in; (void)out_size; (void)d_ws; (void)ws_size;
  const int B = in_sizes[0] / 40;                 // past is (B, 20, 2)
  dim3 grid(B / 16), block(128);
  encdec_kernel<<<grid, block, SMEM_BYTES, stream>>>(
      (const float*)d_in[0],  (const float*)d_in[1],
      (const float*)d_in[2],  (const float*)d_in[3],
      (const float*)d_in[4],  (const float*)d_in[5],
      (const float*)d_in[6],  (const float*)d_in[7],
      (const float*)d_in[8],  (const float*)d_in[9],
      (const float*)d_in[10], (const float*)d_in[11],
      (const float*)d_in[12], (const float*)d_in[13],
      (const float*)d_in[14], (const float*)d_in[15],
      (const float*)d_in[16], (const float*)d_in[17],
      (const float*)d_in[18], (const float*)d_in[19],
      (const float*)d_in[20], (const float*)d_in[21],
      (const float*)d_in[22], (const float*)d_in[23],
      (float*)d_out);
}
